// AdaptiveScaledDotProductAttention_49795850830147
// MI455X (gfx1250) — matrix operand order repaired and checked by reading, not compiled, not run on hardware
//
#include <hip/hip_runtime.h>
#include <hip/hip_bf16.h>

typedef __bf16 bf16;
typedef __attribute__((ext_vector_type(16))) __bf16 v16bf;
typedef __attribute__((ext_vector_type(8)))  __bf16 v8bf;
typedef __attribute__((ext_vector_type(4)))  __bf16 v4bf;
typedef __attribute__((ext_vector_type(8)))  float  v8f;
typedef __attribute__((ext_vector_type(4)))  float  v4f;

#define B_   8
#define NQ_  1024
#define NK_  1024
#define DM_  1024   // H * D_K
#define H_   8
#define DK_  128

static __device__ __forceinline__ v8f zero8() {
  v8f z;
#pragma unroll
  for (int i = 0; i < 8; ++i) z[i] = 0.f;
  return z;
}

static __device__ __forceinline__ v8f wmma_bf16(v16bf a, v16bf b, v8f c) {
  // D = A(16x32 bf16) * B(32x16 bf16) + C(16x16 f32)
  return __builtin_amdgcn_wmma_f32_16x16x32_bf16(false, a, false, b, (short)0, c,
                                                 false, false);
}

static __device__ __forceinline__ v16bf cat8(v8bf lo, v8bf hi) {
  return __builtin_shufflevector(lo, hi, 0, 1, 2, 3, 4, 5, 6, 7, 8, 9, 10, 11,
                                 12, 13, 14, 15);
}

// A-operand fragment: lane(0..15)=row, K split {ko..ko+7} U {ko+16..ko+23}, ko=8*(lane>=16)
template <int STRIDE>
static __device__ __forceinline__ v16bf ld_fragA(const bf16* base, int lane) {
  const int row = lane & 15;
  const int ko  = (lane >> 4) << 3;
  v8bf lo = *(const v8bf*)(base + row * STRIDE + ko);
  v8bf hi = *(const v8bf*)(base + row * STRIDE + ko + 16);
  return cat8(lo, hi);
}

// B-operand fragment: lane(0..15)=col, K split {ks..ks+7} U {ks+16..ks+23}, ks=8*(lane>=16)
template <int STRIDE>
static __device__ __forceinline__ v16bf ld_fragB(const bf16* base, int lane) {
  const int col = lane & 15;
  const int ks  = (lane >> 4) << 3;
  v8bf lo = *(const v8bf*)(base + col * STRIDE + ks);
  v8bf hi = *(const v8bf*)(base + col * STRIDE + ks + 16);
  return cat8(lo, hi);
}

// CDNA5 async global->LDS copy (ASYNCcnt-tracked, no VGPR data path).
// GVS addressing: mem = SADDR(64) + VADDR(32) ; LDS addr from VDST VGPR.
static __device__ __forceinline__ void async_ld_b128(unsigned lds_addr,
                                                     const void* sbase,
                                                     int gbyteoff) {
  asm volatile("global_load_async_to_lds_b128 %0, %1, %2"
               :
               : "v"(lds_addr), "v"(gbyteoff), "s"(sbase)
               : "memory");
}
static __device__ __forceinline__ void wait_async0() {
  asm volatile("s_wait_asynccnt 0" ::: "memory");
}

// ---------------- fp32 -> bf16 weight conversion ----------------
__global__ __launch_bounds__(256) void cvt_f32_bf16(const float* __restrict__ in,
                                                    bf16* __restrict__ out, int n4) {
  int i = blockIdx.x * 256 + threadIdx.x;
  if (i >= n4) return;
  v4f v = ((const v4f*)in)[i];
  v4bf o;
#pragma unroll
  for (int j = 0; j < 4; ++j) o[j] = (bf16)v[j];
  ((v4bf*)out)[i] = o;
}

// ---------------- GEMM: C(M,1024) = A(M,1024) @ W(1024,1024)^T + bias ----------------
// Block tile 256x128, 8 waves (4x2), 64x64 per wave, k-step 32.
template <typename AT, typename OT>
__global__ __launch_bounds__(256) void gemm_x1024(const AT* __restrict__ A,
                                                  const bf16* __restrict__ W,
                                                  const float* __restrict__ bias,
                                                  OT* __restrict__ C) {
  constexpr int K = 1024, N = 1024;
  __shared__ bf16 As[256 * 32];
  __shared__ bf16 Bs[128 * 32];
  const int t = threadIdx.x, lane = t & 31, w = t >> 5;
  const int m0 = blockIdx.x * 256, n0 = blockIdx.y * 128;
  const int rb = (w >> 1) * 64, cb = (w & 1) * 64;
  v8f c[4][4];
#pragma unroll
  for (int i = 0; i < 4; ++i)
#pragma unroll
    for (int j = 0; j < 4; ++j) c[i][j] = zero8();

  for (int kb = 0; kb < K; kb += 32) {
    if constexpr (sizeof(AT) == 4) {
      // stage A: fp32 -> bf16 on the fly (256x32 tile)
#pragma unroll
      for (int it = 0; it < 8; ++it) {
        int e = (t + it * 256) * 4;
        int r = e >> 5, kk = e & 31;
        v4f av = *(const v4f*)(A + (size_t)(m0 + r) * K + kb + kk);
        v4bf o;
#pragma unroll
        for (int j = 0; j < 4; ++j) o[j] = (bf16)av[j];
        *(v4bf*)(As + r * 32 + kk) = o;
      }
    } else {
      // stage A: async global->LDS, bf16 direct
#pragma unroll
      for (int it = 0; it < 4; ++it) {
        int e = (t + it * 256) * 8;
        int r = e >> 5, kk = e & 31;
        async_ld_b128((unsigned)(size_t)(As + r * 32 + kk), A,
                      (int)(((size_t)(m0 + r) * K + kb + kk) * sizeof(bf16)));
      }
    }
    // stage B (weights, 128x32 tile): async global->LDS
#pragma unroll
    for (int it = 0; it < 2; ++it) {
      int e = (t + it * 256) * 8;
      int r = e >> 5, kk = e & 31;
      async_ld_b128((unsigned)(size_t)(Bs + r * 32 + kk), W,
                    (int)(((size_t)(n0 + r) * K + kb + kk) * sizeof(bf16)));
    }
    wait_async0();
    __syncthreads();

    v16bf af[4], bfr[4];
#pragma unroll
    for (int i = 0; i < 4; ++i) af[i] = ld_fragA<32>(As + (rb + i * 16) * 32, lane);
#pragma unroll
    for (int j = 0; j < 4; ++j) bfr[j] = ld_fragB<32>(Bs + (cb + j * 16) * 32, lane);
#pragma unroll
    for (int j = 0; j < 4; ++j)
#pragma unroll
      for (int i = 0; i < 4; ++i) c[i][j] = wmma_bf16(af[i], bfr[j], c[i][j]);
    __syncthreads();
  }

  const int halo = (lane >> 4) << 3;
#pragma unroll
  for (int i = 0; i < 4; ++i) {
#pragma unroll
    for (int j = 0; j < 4; ++j) {
      int gc = n0 + cb + j * 16 + (lane & 15);
      float bv = bias[gc];
#pragma unroll
      for (int ii = 0; ii < 8; ++ii) {
        int gr = m0 + rb + i * 16 + ii + halo;
        C[(size_t)gr * N + gc] = (OT)(c[i][j][ii] + bv);
      }
    }
  }
}

// ---------------- Flash attention with sentinel column ----------------
// grid = B*H*(NQ/128) blocks; 256 threads (8 waves); wave owns 16 q-rows.
__global__ __launch_bounds__(256) void attn_flash(const bf16* __restrict__ Q,
                                                  const bf16* __restrict__ K,
                                                  const bf16* __restrict__ V,
                                                  const bf16* __restrict__ S,
                                                  bf16* __restrict__ O) {
  constexpr float SCALE = 0.08838834764831845f;  // 1/sqrt(128)
  __shared__ bf16 Ks[32 * 128];      // key tile, row-major (key, d)
  __shared__ bf16 Vt[128 * 32];      // value tile transposed (d, key)
  __shared__ bf16 Ps[8][16 * 32];    // per-wave P scratch (C-layout -> A-layout)

  const int t = threadIdx.x, lane = t & 31, w = t >> 5;
  const int blk = blockIdx.x;
  const int qt = blk & 7, bh = blk >> 3, h = bh & 7, b = bh >> 3;
  const int q0 = qt * 128 + w * 16;
  const int rowA = lane & 15, ko = (lane >> 4) << 3;
  const int halo = (lane >> 4) << 3;

  // persistent Q fragments (A-operand layout) + sentinel dot q_i . s_i
  const bf16* Qrow = Q + ((size_t)(b * NQ_ + q0 + rowA)) * DM_ + h * DK_;
  const bf16* Srow = S + ((size_t)(b * NQ_ + q0 + rowA)) * DM_ + h * DK_;
  v16bf qf[4];
  float sd = 0.f;
  {
#pragma unroll
    for (int f = 0; f < 4; ++f) {
      qf[f] = cat8(*(const v8bf*)(Qrow + f * 32 + ko),
                   *(const v8bf*)(Qrow + f * 32 + ko + 16));
      v16bf sf = cat8(*(const v8bf*)(Srow + f * 32 + ko),
                      *(const v8bf*)(Srow + f * 32 + ko + 16));
#pragma unroll
      for (int e = 0; e < 16; ++e) sd += (float)qf[f][e] * (float)sf[e];
    }
  }
  sd += __shfl_xor(sd, 16, 32);  // lane pair (l, l+16) covers all 128 dims
  sd *= SCALE;

  float m[8], l[8];
  v8f acc[8];
#pragma unroll
  for (int i = 0; i < 8; ++i) { m[i] = -1e30f; l[i] = 0.f; acc[i] = zero8(); }

  for (int kc = 0; kc < NK_; kc += 32) {
    // stage K tile 32x128 row-major via async global->LDS
#pragma unroll
    for (int it = 0; it < 2; ++it) {
      int e = (t + it * 256) * 8;
      int kr = e >> 7, d = e & 127;
      async_ld_b128(
          (unsigned)(size_t)(Ks + kr * 128 + d), K,
          (int)((((size_t)(b * NK_ + kc + kr)) * DM_ + h * DK_ + d) * sizeof(bf16)));
    }
    // stage V tile transposed: Vt[d][k] (register-side transpose)
#pragma unroll
    for (int it = 0; it < 2; ++it) {
      int e = (t + it * 256) * 8;
      int kr = e >> 7, d = e & 127;
      v8bf vv =
          *(const v8bf*)(V + ((size_t)(b * NK_ + kc + kr)) * DM_ + h * DK_ + d);
#pragma unroll
      for (int j = 0; j < 8; ++j) Vt[(d + j) * 32 + kr] = vv[j];
    }
    if (kc + 32 < NK_) {
      const bf16* nk =
          K + ((size_t)(b * NK_ + kc + 32 + (t >> 3))) * DM_ + h * DK_ + (t & 7) * 16;
      __builtin_prefetch(nk, 0, 1);
    }
    wait_async0();
    __syncthreads();

    // scores: 16(q) x 32(k). Load all 8 K-fragments first so the WMMA burst
    // is not serialized behind per-fragment s_wait_dscnt drains.
    v16bf kf[8];
#pragma unroll
    for (int f = 0; f < 4; ++f) {
      kf[2 * f]     = ld_fragB<128>(Ks + f * 32, lane);
      kf[2 * f + 1] = ld_fragB<128>(Ks + 16 * 128 + f * 32, lane);
    }
    v8f s0 = zero8(), s1 = zero8();
#pragma unroll
    for (int f = 0; f < 4; ++f) {
      s0 = wmma_bf16(qf[f], kf[2 * f], s0);
      s1 = wmma_bf16(qf[f], kf[2 * f + 1], s1);
    }

    // online softmax over these 32 keys
    float cm[8], rs[8], al[8];
#pragma unroll
    for (int i = 0; i < 8; ++i) {
      s0[i] *= SCALE;
      s1[i] *= SCALE;
      cm[i] = fmaxf(s0[i], s1[i]);
    }
#pragma unroll
    for (int msk = 1; msk < 16; msk <<= 1) {
#pragma unroll
      for (int i = 0; i < 8; ++i)
        cm[i] = fmaxf(cm[i], __shfl_xor(cm[i], msk, 32));
    }
#pragma unroll
    for (int i = 0; i < 8; ++i) {
      float mn = fmaxf(m[i], cm[i]);
      al[i] = __expf(m[i] - mn);
      float p0 = __expf(s0[i] - mn);
      float p1 = __expf(s1[i] - mn);
      rs[i] = p0 + p1;
      m[i] = mn;
      int prow = i + halo;  // C-layout row of slot i
      Ps[w][prow * 32 + (lane & 15)] = (bf16)p0;
      Ps[w][prow * 32 + 16 + (lane & 15)] = (bf16)p1;
    }
#pragma unroll
    for (int msk = 1; msk < 16; msk <<= 1) {
#pragma unroll
      for (int i = 0; i < 8; ++i) rs[i] += __shfl_xor(rs[i], msk, 32);
    }
#pragma unroll
    for (int i = 0; i < 8; ++i) {
      l[i] = l[i] * al[i] + rs[i];
#pragma unroll
      for (int j = 0; j < 8; ++j) acc[j][i] *= al[i];
    }

    // P (bf16, A-layout via per-wave LDS scratch) @ V-tile, fragments batched 4-wide
    v16bf pf = ld_fragA<32>(&Ps[w][0], lane);
#pragma unroll
    for (int jj = 0; jj < 2; ++jj) {
      v16bf vf[4];
#pragma unroll
      for (int j = 0; j < 4; ++j)
        vf[j] = ld_fragB<32>(Vt + ((jj * 4 + j) * 16) * 32, lane);
#pragma unroll
      for (int j = 0; j < 4; ++j)
        acc[jj * 4 + j] = wmma_bf16(pf, vf[j], acc[jj * 4 + j]);
    }
    __syncthreads();
  }

  // fold in the sentinel column and write out
  float wsent[8];
#pragma unroll
  for (int i = 0; i < 8; ++i) {
    float z = __shfl(sd, i + halo, 32);  // sentinel score for this slot's row
    float mn = fmaxf(m[i], z);
    float a = __expf(m[i] - mn);
    wsent[i] = __expf(z - mn);
    l[i] = l[i] * a + wsent[i];
#pragma unroll
    for (int j = 0; j < 8; ++j) acc[j][i] *= a;
  }
#pragma unroll
  for (int j = 0; j < 8; ++j) {
    int d = j * 16 + (lane & 15);
#pragma unroll
    for (int i = 0; i < 8; ++i) {
      int r = q0 + i + halo;
      size_t off = ((size_t)(b * NQ_ + r)) * DM_ + h * DK_ + d;
      float sval = (float)S[off];
      float o = (acc[j][i] + wsent[i] * sval) / l[i];
      O[off] = (bf16)o;
    }
  }
}

// ---------------- host launch ----------------
extern "C" void kernel_launch(void* const* d_in, const int* in_sizes, int n_in,
                              void* d_out, int out_size, void* d_ws, size_t ws_size,
                              hipStream_t stream) {
  (void)in_sizes; (void)n_in; (void)out_size; (void)ws_size;
  const float* Xq = (const float*)d_in[0];
  const float* Xk = (const float*)d_in[1];
  const float* Xv = (const float*)d_in[2];
  const float* Xs = (const float*)d_in[3];
  const float* Wq = (const float*)d_in[4];
  const float* bq = (const float*)d_in[5];
  const float* Wk = (const float*)d_in[6];
  const float* bk = (const float*)d_in[7];
  const float* Wv = (const float*)d_in[8];
  const float* bv = (const float*)d_in[9];
  const float* Ws = (const float*)d_in[10];
  const float* bs = (const float*)d_in[11];
  const float* Wo = (const float*)d_in[12];
  const float* bo = (const float*)d_in[13];

  bf16* wsb = (bf16*)d_ws;
  const size_t WE = 1024ull * 1024ull;            // weight elems
  const size_t PE = 8ull * 1024ull * 1024ull;     // projected activation elems
  bf16* Wqb = wsb + 0 * WE;
  bf16* Wkb = wsb + 1 * WE;
  bf16* Wvb = wsb + 2 * WE;
  bf16* Wsb = wsb + 3 * WE;
  bf16* Wob = wsb + 4 * WE;
  bf16* Qp = wsb + 5 * WE;
  bf16* Kp = Qp + PE;
  bf16* Vp = Kp + PE;
  bf16* Sp = Vp + PE;
  bf16* AO = Sp + PE;

  const int n4 = (int)(WE / 4);
  cvt_f32_bf16<<<1024, 256, 0, stream>>>(Wq, Wqb, n4);
  cvt_f32_bf16<<<1024, 256, 0, stream>>>(Wk, Wkb, n4);
  cvt_f32_bf16<<<1024, 256, 0, stream>>>(Wv, Wvb, n4);
  cvt_f32_bf16<<<1024, 256, 0, stream>>>(Ws, Wsb, n4);
  cvt_f32_bf16<<<1024, 256, 0, stream>>>(Wo, Wob, n4);

  dim3 gg(32, 8);  // M/256 x N/128
  gemm_x1024<float, bf16><<<gg, 256, 0, stream>>>(Xq, Wqb, bq, Qp);
  gemm_x1024<float, bf16><<<gg, 256, 0, stream>>>(Xk, Wkb, bk, Kp);
  gemm_x1024<float, bf16><<<gg, 256, 0, stream>>>(Xv, Wvb, bv, Vp);
  gemm_x1024<float, bf16><<<gg, 256, 0, stream>>>(Xs, Wsb, bs, Sp);

  attn_flash<<<B_ * H_ * (NQ_ / 128), 256, 0, stream>>>(Qp, Kp, Vp, Sp, AO);

  gemm_x1024<bf16, float><<<gg, 256, 0, stream>>>(AO, Wob, bo, (float*)d_out);
}
